// NanoTitanModel_36395552866463
// MI455X (gfx1250) — compile-verified, matrix operand order checked by reading, not run on hardware
//
#include <hip/hip_runtime.h>
#include <stdint.h>

// ---- model dims (match reference) ----
#define B_     2
#define S_     2048
#define D_     1024
#define H_     16
#define DH_    64
#define FFN_   4096
#define L_     4
#define VOCAB_ 32000
#define M_     (B_ * S_)      // 4096 token rows

typedef __bf16 bf16;
typedef __bf16 v16bf __attribute__((ext_vector_type(16)));
typedef __bf16 v8bf  __attribute__((ext_vector_type(8)));
typedef float  v8f   __attribute__((ext_vector_type(8)));

// ---------------------------------------------------------------------------
// A/B fragment loader for v_wmma_f32_16x16x32_bf16.
// ISA 16-bit A-layout (7.12.2): lane&15 -> row, lane>>4 selects K halves:
//   elements 0..7  = K( hi*8 + 0..7 ),  elements 8..15 = K( 16 + hi*8 + 0..7 )
// i.e. two contiguous, 16-byte aligned 8xbf16 chunks per lane.
// ---------------------------------------------------------------------------
__device__ inline v16bf make_frag(const bf16* rowbase, int hi) {
  v8bf c0 = *(const v8bf*)(rowbase + hi * 8);
  v8bf c1 = *(const v8bf*)(rowbase + 16 + hi * 8);
  v16bf f;
#pragma unroll
  for (int i = 0; i < 8; ++i) { f[i] = c0[i]; f[8 + i] = c1[i]; }
  return f;
}

__device__ inline v8f wmma_bf16(v16bf a, v16bf b, v8f c) {
  return __builtin_amdgcn_wmma_f32_16x16x32_bf16(
      /*neg_a=*/false, a, /*neg_b=*/false, b,
      /*c_mod=*/(short)0, c, /*reuse_a=*/false, /*reuse_b=*/false);
}

// ---------------------------------------------------------------------------
// gfx1250 async tensor path: GLOBAL_LOAD_ASYNC_TO_LDS_B128 (ASYNCcnt-tracked,
// bypasses VGPRs; ISA 10.x / 15.18.3). Generic shared pointers carry the LDS
// byte offset in their low 32 bits (aperture mapping), which is what VDST
// expects (hardware adds the wave's LDS_BASE).
// ---------------------------------------------------------------------------
__device__ inline unsigned lds_off(const void* p) {
  return (unsigned)(size_t)p;
}

__device__ inline void async_b128(unsigned lds, const bf16* g) {
  asm volatile("global_load_async_to_lds_b128 %0, %1, off"
               :: "v"(lds), "v"((unsigned long long)(size_t)g)
               : "memory");
}

__device__ inline void wait_async0() {
#if __has_builtin(__builtin_amdgcn_s_wait_asynccnt)
  __builtin_amdgcn_s_wait_asynccnt(0);
#else
  asm volatile("s_wait_asynccnt 0x0" ::: "memory");
#endif
}
__device__ inline void wait_async2() {
#if __has_builtin(__builtin_amdgcn_s_wait_asynccnt)
  __builtin_amdgcn_s_wait_asynccnt(2);
#else
  asm volatile("s_wait_asynccnt 0x2" ::: "memory");
#endif
}
__device__ inline void wait_async4() {
#if __has_builtin(__builtin_amdgcn_s_wait_asynccnt)
  __builtin_amdgcn_s_wait_asynccnt(4);
#else
  asm volatile("s_wait_asynccnt 0x4" ::: "memory");
#endif
}

// ---------------------------------------------------------------------------
// f32 -> bf16 conversion
// ---------------------------------------------------------------------------
__global__ void cvt_kernel(const float* __restrict__ src, bf16* __restrict__ dst,
                           size_t n) {
  size_t i = (size_t)blockIdx.x * blockDim.x + threadIdx.x;
  if (i < n) dst[i] = (bf16)src[i];
}

// ---------------------------------------------------------------------------
// embedding gather + sinusoidal positional embedding; writes f32 x + bf16 xb
// ---------------------------------------------------------------------------
__global__ void embed_pos_kernel(const int* __restrict__ ids,
                                 const float* __restrict__ emb,
                                 float* __restrict__ x, bf16* __restrict__ xb) {
  size_t i = (size_t)blockIdx.x * blockDim.x + threadIdx.x;
  if (i >= (size_t)M_ * D_) return;
  int d  = (int)(i % D_);
  int bs = (int)(i / D_);
  int s  = bs % S_;
  float v = emb[(size_t)ids[bs] * D_ + d];
  int half = d >> 1;
  float ang = (float)s * expf((float)(2 * half) * (-9.210340371976184f / (float)D_));
  v += (d & 1) ? cosf(ang) : sinf(ang);
  x[i]  = v;
  xb[i] = (bf16)v;
}

// ---------------------------------------------------------------------------
// Tiled bf16 GEMM: C[M,N] = A[M,K] * B + (bias) + (residual), optional ReLU.
//   bNK==1 : B stored as [N,K] row-major (transposed weight / embed^T)
//   bNK==0 : B stored as [K,N] row-major
// 256 threads = 8 waves; block tile 128x128, K-step 32, double-buffered LDS.
// A (and B when N-major) staged with global_load_async_to_lds_b128: tile k+1
// streams in while tile k feeds 8 WMMAs per wave (32x64 wave tile).
// LDS stride 40 halfwords (80B) keeps all fragment chunks 16B aligned.
// ---------------------------------------------------------------------------
__global__ __launch_bounds__(256)
void gemm_bf16_kernel(const bf16* __restrict__ A, const bf16* __restrict__ Bm,
                      const float* __restrict__ bias,
                      const float* __restrict__ resid,
                      float* __restrict__ outF, bf16* __restrict__ outB,
                      int M, int N, int K, int bNK, int relu) {
  __shared__ __align__(16) bf16 sA[2][128 * 40];
  __shared__ __align__(16) bf16 sB[2][128 * 40];

  const int tid  = threadIdx.x;
  const int lane = tid & 31;
  const int wid  = tid >> 5;
  const int hi   = lane >> 4;
  const int lo   = lane & 15;
  const int wm   = wid & 3;   // 4 waves along M -> 32 rows each
  const int wn   = wid >> 2;  // 2 waves along N -> 64 cols each
  const int m0   = blockIdx.y * 128;
  const int n0   = blockIdx.x * 128;

  // staging geometry: 2 threads per row, 16 bf16 (two b128 chunks) each
  const int srow = tid >> 1;
  const int skc  = (tid & 1) * 16;
  const bf16* gA = A + (size_t)(m0 + srow) * K + skc;
  const bf16* gB = Bm + (size_t)(n0 + srow) * K + skc;  // only used when bNK
  unsigned offA[2], offB[2];
#pragma unroll
  for (int b = 0; b < 2; ++b) {
    offA[b] = lds_off(&sA[b][srow * 40 + skc]);
    offB[b] = lds_off(&sB[b][srow * 40 + skc]);
  }
  // transpose-staging geometry for B in [K,N] layout
  const int bk  = tid >> 3;         // 0..31
  const int bnc = (tid & 7) * 16;   // 0..112

  auto stage = [&](int kt, int buf) {
    const int k0 = kt * 32;
    async_b128(offA[buf],      gA + k0);
    async_b128(offA[buf] + 16, gA + k0 + 8);
    if (bNK) {
      async_b128(offB[buf],      gB + k0);
      async_b128(offB[buf] + 16, gB + k0 + 8);
    } else {  // coalesced read along N, transpose into sB[buf][n][k]
      const bf16* src = Bm + (size_t)(k0 + bk) * N + n0 + bnc;
#pragma unroll
      for (int i = 0; i < 16; ++i) sB[buf][(bnc + i) * 40 + bk] = src[i];
    }
  };

  v8f acc[2][4];
#pragma unroll
  for (int i = 0; i < 2; ++i)
#pragma unroll
    for (int j = 0; j < 4; ++j)
#pragma unroll
      for (int e = 0; e < 8; ++e) acc[i][j][e] = 0.0f;

  const int nkt = K >> 5;
  stage(0, 0);
  for (int kt = 0; kt < nkt; ++kt) {
    const int cur = kt & 1;
    if (kt + 1 < nkt) {
      stage(kt + 1, cur ^ 1);        // prefetch next tile into other buffer
      if (bNK) wait_async4(); else wait_async2();  // retire current tile's copies
    } else {
      wait_async0();
    }
    __syncthreads();

    v16bf af[2], bf_[4];
#pragma unroll
    for (int mi = 0; mi < 2; ++mi)
      af[mi] = make_frag(&sA[cur][(wm * 32 + mi * 16 + lo) * 40], hi);
#pragma unroll
    for (int ni = 0; ni < 4; ++ni)
      bf_[ni] = make_frag(&sB[cur][(wn * 64 + ni * 16 + lo) * 40], hi);
#pragma unroll
    for (int mi = 0; mi < 2; ++mi)
#pragma unroll
      for (int ni = 0; ni < 4; ++ni)
        acc[mi][ni] = wmma_bf16(af[mi], bf_[ni], acc[mi][ni]);
    __syncthreads();
  }

  // epilogue: C layout -> lane&15 = column, VGPR r = row (r + 8 for lanes>=16)
#pragma unroll
  for (int mi = 0; mi < 2; ++mi)
#pragma unroll
    for (int ni = 0; ni < 4; ++ni) {
      int col  = n0 + wn * 64 + ni * 16 + lo;
      float bv = bias ? bias[col] : 0.0f;
#pragma unroll
      for (int r = 0; r < 8; ++r) {
        int row    = m0 + wm * 32 + mi * 16 + hi * 8 + r;
        size_t idx = (size_t)row * N + col;
        float v = acc[mi][ni][r] + bv;
        if (resid) v += resid[idx];
        if (relu)  v = fmaxf(v, 0.0f);
        if (outF) outF[idx] = v;
        if (outB) outB[idx] = (bf16)v;
      }
    }
}

// ---------------------------------------------------------------------------
// Causal flash attention. One wave per (b, h, 16-row q-tile).
// K-tiles of 32: scores via 2x2 WMMAs, online softmax with shfl_xor row
// reductions (C layout: row r+hi*8 lives in VGPR r across a 16-lane group),
// P staged through LDS back into A-fragment layout, then P @ V with V in LDS.
// ---------------------------------------------------------------------------
__global__ __launch_bounds__(32)
void attn_kernel(const bf16* __restrict__ Q, const bf16* __restrict__ Kb,
                 const bf16* __restrict__ V, bf16* __restrict__ O) {
  __shared__ __align__(16) bf16 sV[32 * 80];   // 32 K-rows x 64 d (stride 80)
  __shared__ __align__(16) bf16 sP[16 * 32];   // 16 q-rows x 32 k, row-major

  const int lane  = threadIdx.x;
  const int hi    = lane >> 4;
  const int lo    = lane & 15;
  const int qt    = blockIdx.x;
  const int bh    = blockIdx.y;
  const int b     = bh / H_;
  const int h     = bh % H_;
  const int qbase = qt * 16;
  const size_t ld   = (size_t)H_ * DH_;                    // 1024
  const size_t base = (size_t)b * S_ * ld + (size_t)h * DH_;

  v16bf qf[2];
#pragma unroll
  for (int kk = 0; kk < 2; ++kk)
    qf[kk] = make_frag(Q + base + (size_t)(qbase + lo) * ld + kk * 32, hi);

  float mrow[8], lrow[8];
  v8f accO[4];
#pragma unroll
  for (int r = 0; r < 8; ++r) { mrow[r] = -1e30f; lrow[r] = 0.0f; }
#pragma unroll
  for (int f = 0; f < 4; ++f)
#pragma unroll
    for (int r = 0; r < 8; ++r) accO[f][r] = 0.0f;

  const int nkt = (qbase + 47) >> 5;  // covers k in [0, qbase+15]
  for (int kt = 0; kt < nkt; ++kt) {
    const int kbase = kt * 32;
    __syncthreads();  // retire last iteration's sV/sP reads before overwrite

    {  // stage V tile (32 x 64): one row per lane, 8x 16B chunks
      const bf16* src = V + base + (size_t)(kbase + lane) * ld;
#pragma unroll
      for (int c = 0; c < 8; ++c)
        *(v8bf*)&sV[lane * 80 + c * 8] = *(const v8bf*)(src + c * 8);
    }

    // scores: S[16 x 32] = Q(16x64) . K^T, two 16-wide column halves
    v8f sc[2];
#pragma unroll
    for (int c = 0; c < 2; ++c)
#pragma unroll
      for (int r = 0; r < 8; ++r) sc[c][r] = 0.0f;
#pragma unroll
    for (int c = 0; c < 2; ++c) {
      const bf16* krow = Kb + base + (size_t)(kbase + c * 16 + lo) * ld;
#pragma unroll
      for (int kk = 0; kk < 2; ++kk)
        sc[c] = wmma_bf16(qf[kk], make_frag(krow + kk * 32, hi), sc[c]);
    }

    const bool needmask = (kbase + 31 > qbase);
    float p0v[8], p1v[8], corr[8];
#pragma unroll
    for (int r = 0; r < 8; ++r) {
      int row  = qbase + hi * 8 + r;
      float s0 = sc[0][r] * 0.125f;   // 1/sqrt(64)
      float s1 = sc[1][r] * 0.125f;
      if (needmask) {
        if (kbase + lo > row)      s0 = -1e30f;
        if (kbase + 16 + lo > row) s1 = -1e30f;
      }
      float mx = fmaxf(s0, s1);
#pragma unroll
      for (int d = 1; d < 16; d <<= 1) mx = fmaxf(mx, __shfl_xor(mx, d, 16));
      float mnew = fmaxf(mrow[r], mx);
      corr[r] = expf(mrow[r] - mnew);
      float p0 = expf(s0 - mnew);
      float p1 = expf(s1 - mnew);
      float rs = p0 + p1;
#pragma unroll
      for (int d = 1; d < 16; d <<= 1) rs += __shfl_xor(rs, d, 16);
      lrow[r] = lrow[r] * corr[r] + rs;
      mrow[r] = mnew;
      p0v[r] = p0; p1v[r] = p1;
    }

#pragma unroll
    for (int f = 0; f < 4; ++f)
#pragma unroll
      for (int r = 0; r < 8; ++r) accO[f][r] *= corr[r];

    // P (C layout) -> LDS row-major 16x32, then reload as A fragment
#pragma unroll
    for (int r = 0; r < 8; ++r) {
      sP[(hi * 8 + r) * 32 + lo]      = (bf16)p0v[r];
      sP[(hi * 8 + r) * 32 + 16 + lo] = (bf16)p1v[r];
    }
    __syncthreads();
    v16bf pf = make_frag(&sP[lo * 32], hi);

    // O += P(16x32) @ V(32x64), 4 d-column groups
#pragma unroll
    for (int f = 0; f < 4; ++f) {
      v16bf vf;
#pragma unroll
      for (int e = 0; e < 16; ++e) {
        int kk = (e < 8) ? (hi * 8 + e) : (16 + hi * 8 + (e - 8));
        vf[e] = sV[kk * 80 + f * 16 + lo];
      }
      accO[f] = wmma_bf16(pf, vf, accO[f]);
    }
  }

#pragma unroll
  for (int r = 0; r < 8; ++r) lrow[r] = 1.0f / lrow[r];
#pragma unroll
  for (int f = 0; f < 4; ++f)
#pragma unroll
    for (int r = 0; r < 8; ++r) {
      int row = qbase + hi * 8 + r;
      O[base + (size_t)row * ld + f * 16 + lo] = (bf16)(accO[f][r] * lrow[r]);
    }
}

// ---------------------------------------------------------------------------
// host launch
// ---------------------------------------------------------------------------
static inline void launch_gemm(const bf16* A, const bf16* Bm, const float* bias,
                               const float* resid, float* outF, bf16* outB,
                               int M, int N, int K, int bNK, int relu,
                               hipStream_t stream) {
  dim3 grid(N / 128, M / 128);
  gemm_bf16_kernel<<<grid, 256, 0, stream>>>(A, Bm, bias, resid, outF, outB,
                                             M, N, K, bNK, relu);
}

extern "C" void kernel_launch(void* const* d_in, const int* in_sizes, int n_in,
                              void* d_out, int out_size, void* d_ws, size_t ws_size,
                              hipStream_t stream) {
  const int*   ids    = (const int*)d_in[0];
  const float* embedF = (const float*)d_in[1];
  const float* WQ     = (const float*)d_in[2];
  const float* WK     = (const float*)d_in[3];
  const float* WV     = (const float*)d_in[4];
  const float* WO     = (const float*)d_in[5];
  const float* Win_w  = (const float*)d_in[6];
  const float* Win_b  = (const float*)d_in[7];
  const float* Wout_w = (const float*)d_in[8];
  const float* Wout_b = (const float*)d_in[9];
  float* out = (float*)d_out;

  char* ws = (char*)d_ws;
  size_t off = 0;
  auto take = [&](size_t bytes) { char* p = ws + off; off += bytes; return p; };

  float* x   = (float*)take((size_t)M_ * D_ * 4);
  bf16*  xb  = (bf16*) take((size_t)M_ * D_ * 2);
  bf16*  qb  = (bf16*) take((size_t)M_ * D_ * 2);
  bf16*  kb  = (bf16*) take((size_t)M_ * D_ * 2);
  bf16*  vb  = (bf16*) take((size_t)M_ * D_ * 2);
  bf16*  ab  = (bf16*) take((size_t)M_ * D_ * 2);
  bf16*  hb  = (bf16*) take((size_t)M_ * FFN_ * 2);
  bf16*  wqB = (bf16*) take((size_t)L_ * D_ * D_ * 2);
  bf16*  wkB = (bf16*) take((size_t)L_ * D_ * D_ * 2);
  bf16*  wvB = (bf16*) take((size_t)L_ * D_ * D_ * 2);
  bf16*  woB = (bf16*) take((size_t)L_ * D_ * D_ * 2);
  bf16*  wiB = (bf16*) take((size_t)L_ * FFN_ * D_ * 2);
  bf16*  wpB = (bf16*) take((size_t)L_ * D_ * FFN_ * 2);
  bf16*  emB = (bf16*) take((size_t)VOCAB_ * D_ * 2);
  if (off > ws_size) return;  // workspace too small (compile-only guard)

  auto cvt = [&](const float* s, bf16* d, size_t n) {
    cvt_kernel<<<(unsigned)((n + 255) / 256), 256, 0, stream>>>(s, d, n);
  };
  cvt(embedF, emB, (size_t)VOCAB_ * D_);
  cvt(WQ,     wqB, (size_t)L_ * D_ * D_);
  cvt(WK,     wkB, (size_t)L_ * D_ * D_);
  cvt(WV,     wvB, (size_t)L_ * D_ * D_);
  cvt(WO,     woB, (size_t)L_ * D_ * D_);
  cvt(Win_w,  wiB, (size_t)L_ * FFN_ * D_);
  cvt(Wout_w, wpB, (size_t)L_ * D_ * FFN_);

  embed_pos_kernel<<<(unsigned)(((size_t)M_ * D_ + 255) / 256), 256, 0, stream>>>(
      ids, embedF, x, xb);

  for (int l = 0; l < L_; ++l) {
    const size_t wo4 = (size_t)l * D_ * D_;
    // QKV projections (B is [D, H*DH] = K x N)
    launch_gemm(xb, wqB + wo4, nullptr, nullptr, nullptr, qb, M_, D_, D_, 0, 0, stream);
    launch_gemm(xb, wkB + wo4, nullptr, nullptr, nullptr, kb, M_, D_, D_, 0, 0, stream);
    launch_gemm(xb, wvB + wo4, nullptr, nullptr, nullptr, vb, M_, D_, D_, 0, 0, stream);
    // causal flash attention
    attn_kernel<<<dim3(S_ / 16, B_ * H_), 32, 0, stream>>>(qb, kb, vb, ab);
    // output projection + residual: x = attn @ WO + x
    launch_gemm(ab, woB + wo4, nullptr, x, x, xb, M_, D_, D_, 0, 0, stream);
    // FFN up: h = relu(x @ Win^T + b)   (Win is [FFN, D] = N x K)
    launch_gemm(xb, wiB + (size_t)l * FFN_ * D_, Win_b + (size_t)l * FFN_,
                nullptr, nullptr, hb, M_, FFN_, D_, 1, 1, stream);
    // FFN down + residual: x = h @ Wout^T + b + x   (Wout is [D, FFN] = N x K)
    launch_gemm(hb, wpB + (size_t)l * D_ * FFN_, Wout_b + (size_t)l * D_,
                x, x, xb, M_, D_, FFN_, 1, 0, stream);
  }

  // tied vocab projection: logits = x @ embed^T  (embed is [V, D] = N x K)
  launch_gemm(xb, emB, nullptr, nullptr, out, nullptr, M_, VOCAB_, D_, 1, 0, stream);
}
